// LPGMISTA_72430328480038
// MI455X (gfx1250) — compile-verified, matrix operand order checked by reading, not run on hardware
//
#include <hip/hip_runtime.h>
#include <hip/hip_bf16.h>

// LPG-MISTA fused kernel for gfx1250 (MI455X).
// One workgroup = 16 rows of the batch, fully LDS-resident:
//   - W_y / W_x staged in LDS as f16 (128 KB, reused buffer)
//   - GEMMs via v_wmma_f32_16x16x32_f16 (wave j owns the 16-col output strip j)
//   - Wy_y kept purely in registers (v8f per wave) as the accumulator seed
//   - TV-L1 prox: one wave per row, 100 FISTA dual steps, cross-lane via shfl
//   - FISTA momentum coefficients precomputed at COMPILE TIME (constexpr) so the
//     800-step serial chain has no sqrt/div, just one uniform scalar load/step

typedef __attribute__((ext_vector_type(16))) _Float16 v16h;
typedef __attribute__((ext_vector_type(8)))  _Float16 v8h;
typedef __attribute__((ext_vector_type(4)))  _Float16 v4h;
typedef __attribute__((ext_vector_type(8)))  float    v8f;

#define LMBD_1   0.1f
#define LMBD_2   0.1f
#define MAX_ITER 8
#define TV_ITERS 100

constexpr int MDIM = 128;   // m
constexpr int NDIM = 256;   // n
constexpr int ROWS = 16;    // batch rows per workgroup
constexpr int NT   = 512;   // threads per workgroup (16 waves)

// ---- compile-time FISTA beta_k = (t_k - 1)/t_{k+1} table --------------------
constexpr double csqrt_newton(double x) {
    double r = x > 1.0 ? x : 1.0;
    for (int i = 0; i < 64; ++i) r = 0.5 * (r + x / r);
    return r;
}
struct BetaTab { float b[TV_ITERS]; };
constexpr BetaTab make_beta() {
    BetaTab tab{};
    double tk = 1.0;
    for (int s = 0; s < TV_ITERS; ++s) {
        double tn = 0.5 * (1.0 + csqrt_newton(1.0 + 4.0 * tk * tk));
        tab.b[s] = (float)((tk - 1.0) / tn);
        tk = tn;
    }
    return tab;
}
__constant__ BetaTab BETA = make_beta();

__device__ __forceinline__ float softf(float x, float th) {
    float a = fmaxf(fabsf(x) - th, 0.0f);
    return copysignf(a, x);
}

// Load one 16x32 f16 WMMA operand fragment (A or B) for this lane.
// rowptr = start of this lane's matrix row (A: row M=lane%16 of X;
// B: row j of W, i.e. column j of W^T). kb = kk*32 + (lane>>4)*8.
// Element e of the v16h holds K = (e>>3)*16 + (lane>>4)*8 + (e&7).
__device__ __forceinline__ v16h load_frag16(const _Float16* rowptr, int kb) {
    v8h c0 = *(const v8h*)(rowptr + kb);
    v8h c1 = *(const v8h*)(rowptr + kb + 16);
    v16h a;
#pragma unroll
    for (int i = 0; i < 8; ++i) { a[i] = c0[i]; a[i + 8] = c1[i]; }
    return a;
}

// Row-wise prox_{lam*TV}(x) : one wave handles one row of 256 floats.
// Dual z has 255 entries; the invalid slot (index 255, lane 31 / e==7) is
// pinned to zero by clamping it to [-0,0] via lamHi = 0 on lane 31.
__device__ __forceinline__ void tv_prox_row(float* __restrict__ xrow,
                                            _Float16* __restrict__ xhrow,
                                            int lane, float lam) {
    float xloc[8];
    {
        float4 f0 = *(const float4*)(xrow + lane * 8);
        float4 f1 = *(const float4*)(xrow + lane * 8 + 4);
        xloc[0] = f0.x; xloc[1] = f0.y; xloc[2] = f0.z; xloc[3] = f0.w;
        xloc[4] = f1.x; xloc[5] = f1.y; xloc[6] = f1.z; xloc[7] = f1.w;
    }
    const float lamHi = (lane == 31) ? 0.0f : lam;   // clamp bound for e==7

    float zp[8], zy[8];
#pragma unroll
    for (int e = 0; e < 8; ++e) { zp[e] = 0.0f; zy[e] = 0.0f; }

#pragma unroll 2
    for (int s = 0; s < TV_ITERS; ++s) {
        // v = x - Dt(zy)  (Dt with zero padding: v[i] = x[i] - zy[i-1] + zy[i])
        float zy_up = __shfl_up(zy[7], 1);
        zy_up = (lane == 0) ? 0.0f : zy_up;
        float v[8];
        v[0] = xloc[0] - zy_up + zy[0];
#pragma unroll
        for (int e = 1; e < 8; ++e) v[e] = xloc[e] - zy[e - 1] + zy[e];
        float v_dn = __shfl_down(v[0], 1);

        const float beta = BETA.b[s];                // uniform scalar load

#pragma unroll
        for (int e = 0; e < 8; ++e) {
            float vnext = (e == 7) ? v_dn : v[e + 1];
            float g   = v[e] - vnext;                // grad = -(v[i+1]-v[i])
            float zn  = fmaf(-0.25f, g, zy[e]);
            float lim = (e == 7) ? lamHi : lam;
            zn = fminf(lim, fmaxf(-lim, zn));
            zy[e] = fmaf(beta, zn - zp[e], zn);
            zp[e] = zn;
        }
    }

    // result = x - Dt(z_final)
    float zup = __shfl_up(zp[7], 1);
    zup = (lane == 0) ? 0.0f : zup;
    float o[8];
    o[0] = xloc[0] - zup + zp[0];
#pragma unroll
    for (int e = 1; e < 8; ++e) o[e] = xloc[e] - zp[e - 1] + zp[e];

    *(float4*)(xrow + lane * 8)     = make_float4(o[0], o[1], o[2], o[3]);
    *(float4*)(xrow + lane * 8 + 4) = make_float4(o[4], o[5], o[6], o[7]);
    v8h hh;
#pragma unroll
    for (int e = 0; e < 8; ++e) hh[e] = (_Float16)o[e];
    *(v8h*)(xhrow + lane * 8) = hh;
}

__global__ __launch_bounds__(NT, 1)
void lpg_mista_fused(const float* __restrict__ y,
                     const float* __restrict__ Wx,
                     const float* __restrict__ Wy,
                     const float* __restrict__ u_p,
                     const float* __restrict__ t_p,
                     float* __restrict__ out) {
    __shared__ _Float16 Wsh[NDIM * NDIM];   // 128 KB; phase 1 holds W_y (256x128)
    __shared__ float    xs [ROWS * NDIM];   // 16 KB   current x (f32)
    __shared__ _Float16 xh [ROWS * NDIM];   // 8 KB    x as f16 A-operand (phase 1: y tile)

    const int tid  = threadIdx.x;
    const int lane = tid & 31;
    const int wave = tid >> 5;        // 0..15 : output col strip / TV row
    const int hsel = lane >> 4;       // 0/1
    const int l15  = lane & 15;
    const int row0 = blockIdx.x * ROWS;

    // prefetch W_x into L2/L0 now; it is consumed only after the phase-1 GEMM,
    // so the fetch overlaps W_y staging + WMMA (emits global_prefetch_b8)
#pragma unroll
    for (int p = 0; p < 8; ++p)
        __builtin_prefetch(Wx + (size_t)(tid + p * NT) * 16, 0, 1);

    float u_ = u_p[0]; if (u_ <= 0.0f) u_ = 1e-30f;
    float t_ = t_p[0]; if (t_ <= 0.0f) t_ = 1e-30f;
    const float r_  = t_ / u_;
    const float omr = 1.0f - r_;
    const float l1u = LMBD_1 * u_;
    const float lam = LMBD_2 * t_;

    // ---- stage W_y (f32 -> f16) and this block's y tile into LDS ----
    {
        const float4* src = (const float4*)Wy;                 // 256*128 floats
        for (int i = tid; i < (NDIM * MDIM) / 4; i += NT) {
            float4 f = src[i];
            v4h h; h[0] = (_Float16)f.x; h[1] = (_Float16)f.y;
                   h[2] = (_Float16)f.z; h[3] = (_Float16)f.w;
            *(v4h*)(Wsh + i * 4) = h;
        }
        const float4* ys = (const float4*)(y + (size_t)row0 * MDIM);
        for (int i = tid; i < (ROWS * MDIM) / 4; i += NT) {
            float4 f = ys[i];
            v4h h; h[0] = (_Float16)f.x; h[1] = (_Float16)f.y;
                   h[2] = (_Float16)f.z; h[3] = (_Float16)f.w;
            *(v4h*)(xh + i * 4) = h;
        }
    }
    __syncthreads();

    // ---- phase 1: Wy_y tile = y @ W_y^T, kept in registers per wave ----
    v8f wyacc = {};
    {
        const _Float16* arow = xh  + l15 * MDIM;               // y row, stride 128
        const _Float16* brow = Wsh + (wave * 16 + l15) * MDIM; // W_y row j
#pragma unroll
        for (int kk = 0; kk < MDIM / 32; ++kk) {
            int kb = kk * 32 + hsel * 8;
            v16h a = load_frag16(arow, kb);
            v16h b = load_frag16(brow, kb);
            wyacc = __builtin_amdgcn_wmma_f32_16x16x32_f16(
                false, a, false, b, (short)0, wyacc, false, false);
        }
    }
    // x0 = r * soft(Wy_y, l1*u)  (each lane writes its own C-layout slots)
#pragma unroll
    for (int rr = 0; rr < 8; ++rr) {
        int row = rr + 8 * hsel;
        int col = wave * 16 + l15;
        xs[row * NDIM + col] = r_ * softf(wyacc[rr], l1u);
    }
    __syncthreads();   // Wsh reads done; xs complete

    // ---- overwrite LDS with W_x (f32 -> f16), then first TV prox ----
    {
        const float4* src = (const float4*)Wx;                 // 256*256 floats
        for (int i = tid; i < (NDIM * NDIM) / 4; i += NT) {
            float4 f = src[i];
            v4h h; h[0] = (_Float16)f.x; h[1] = (_Float16)f.y;
                   h[2] = (_Float16)f.z; h[3] = (_Float16)f.w;
            *(v4h*)(Wsh + i * 4) = h;
        }
    }
    tv_prox_row(xs + wave * NDIM, xh + wave * NDIM, lane, lam);
    __syncthreads();

    // ---- 7 unrolled ISTA iterations ----
#pragma unroll 1
    for (int it = 0; it < MAX_ITER - 1; ++it) {
        v8f acc = wyacc;                                       // + Wy_y seed
        const _Float16* arow = xh  + l15 * NDIM;               // x row M=lane%16
        const _Float16* brow = Wsh + (wave * 16 + l15) * NDIM; // W_x row (B col)
#pragma unroll
        for (int kk = 0; kk < NDIM / 32; ++kk) {
            int kb = kk * 32 + hsel * 8;
            v16h a = load_frag16(arow, kb);
            v16h b = load_frag16(brow, kb);
            acc = __builtin_amdgcn_wmma_f32_16x16x32_f16(
                false, a, false, b, (short)0, acc, false, false);
        }
        // x = (1-r)*x + r*soft(acc, l1*u)   (same-thread read-modify-write)
#pragma unroll
        for (int rr = 0; rr < 8; ++rr) {
            int row = rr + 8 * hsel;
            int col = wave * 16 + l15;
            float xold = xs[row * NDIM + col];
            xs[row * NDIM + col] = omr * xold + r_ * softf(acc[rr], l1u);
        }
        __syncthreads();
        tv_prox_row(xs + wave * NDIM, xh + wave * NDIM, lane, lam);
        __syncthreads();
    }

    // ---- write final x tile ----
    {
        float4*       dst = (float4*)(out + (size_t)row0 * NDIM);
        const float4* sx  = (const float4*)xs;
        for (int i = tid; i < (ROWS * NDIM) / 4; i += NT) dst[i] = sx[i];
    }
}

extern "C" void kernel_launch(void* const* d_in, const int* in_sizes, int n_in,
                              void* d_out, int out_size, void* d_ws, size_t ws_size,
                              hipStream_t stream) {
    (void)n_in; (void)out_size; (void)d_ws; (void)ws_size;
    const float* y  = (const float*)d_in[0];
    const float* Wx = (const float*)d_in[1];
    const float* Wy = (const float*)d_in[2];
    const float* u  = (const float*)d_in[3];
    const float* t  = (const float*)d_in[4];
    float* out = (float*)d_out;

    const int B = in_sizes[0] / MDIM;     // 8192
    const int blocks = B / ROWS;          // 512 workgroups of 16 waves
    lpg_mista_fused<<<blocks, NT, 0, stream>>>(y, Wx, Wy, u, t, out);
}
